// CausalSelfAttention_7413113552965
// MI455X (gfx1250) — matrix operand order repaired and checked by reading, not compiled, not run on hardware
//
#include <hip/hip_runtime.h>
#include <stdint.h>

#define D_MODEL 2048
#define SEQ     2048
#define BATCH   2
#define NHEAD   16
#define HDIM    128
#define QKV_N   (3 * D_MODEL)
#define MROWS   (BATCH * SEQ)

typedef __attribute__((ext_vector_type(16))) __bf16 v16bf;
typedef __attribute__((ext_vector_type(8)))  float  v8f;

struct B32 { uint4 lo, hi; };   // 32-byte fragment container

static __device__ __forceinline__ v16bf bitcast_frag(B32 d) {
    return __builtin_bit_cast(v16bf, d);
}

// fp32 -> bf16, round-to-nearest-even
static __device__ __forceinline__ uint16_t f32_to_bf16(float f) {
    uint32_t u = __builtin_bit_cast(uint32_t, f);
    u += 0x7FFFu + ((u >> 16) & 1u);
    return (uint16_t)(u >> 16);
}

// A fragment 16x32 (MxK), K contiguous in memory, ld in elements.
// Lane layout (ISA 7.12.2): lanes 0-15 row M=lane, K {0..7,16..23}; lanes 16-31 K {8..15,24..31}.
static __device__ __forceinline__ v16bf load_frag_a(const uint16_t* __restrict__ base,
                                                    int ld, int row0, int k0) {
    int lane = threadIdx.x & 31;
    int r = lane & 15, hi = lane >> 4;
    const uint16_t* p = base + (size_t)(row0 + r) * ld + k0 + hi * 8;
    B32 d;
    d.lo = *reinterpret_cast<const uint4*>(p);        // K = hi*8 .. hi*8+7
    d.hi = *reinterpret_cast<const uint4*>(p + 16);   // K = 16+hi*8 ..
    return bitcast_frag(d);
}

// B fragment 32x16 (KxN) where column n is contiguous along K in memory (NT GEMM).
// Lane layout: lanes 0-15 col N=lane, K {0..7,16..23}; lanes 16-31 col N=lane-16, K {8..15,24..31}.
static __device__ __forceinline__ v16bf load_frag_b(const uint16_t* __restrict__ base,
                                                    int ld, int col0, int k0) {
    int lane = threadIdx.x & 31;
    int c = lane & 15, hi = lane >> 4;
    const uint16_t* p = base + (size_t)(col0 + c) * ld + k0 + hi * 8;
    B32 d;
    d.lo = *reinterpret_cast<const uint4*>(p);        // K = hi*8 .. hi*8+7
    d.hi = *reinterpret_cast<const uint4*>(p + 16);   // K = 16+hi*8 ..
    return bitcast_frag(d);
}

#define WMMA_BF16(a, b, c) \
    __builtin_amdgcn_wmma_f32_16x16x32_bf16(false, (a), false, (b), (short)0, (c), false, false)

// ---------------- fp32 -> bf16 conversion ----------------
__global__ void cvt_f32_bf16_kernel(const float4* __restrict__ src,
                                    ushort4* __restrict__ dst, int n4) {
    int i = blockIdx.x * blockDim.x + threadIdx.x;
    if (i >= n4) return;
    float4 v = src[i];
    ushort4 o;
    o.x = f32_to_bf16(v.x); o.y = f32_to_bf16(v.y);
    o.z = f32_to_bf16(v.z); o.w = f32_to_bf16(v.w);
    dst[i] = o;
}

// ---------------- QKV projection: qkv = x @ Wqkv^T + b ----------------
// Wave tile: 32M x 64N. Writes Q,K as (B,H,S,Hd) bf16; V transposed as (B,H,Hd,S) bf16.
__global__ __launch_bounds__(256) void qkv_gemm_kernel(
    const uint16_t* __restrict__ xbf, const uint16_t* __restrict__ wbf,
    const float* __restrict__ bias,
    uint16_t* __restrict__ Qbf, uint16_t* __restrict__ Kbf, uint16_t* __restrict__ Vtbf)
{
    int wave = blockIdx.x * (blockDim.x >> 5) + (threadIdx.x >> 5);
    const int NT = QKV_N / 64;
    int mt = wave / NT, nt = wave % NT;
    int m0 = mt * 32, n0 = nt * 64;

    v8f acc0[4] = {};   // rows m0..m0+15
    v8f acc1[4] = {};   // rows m0+16..m0+31
    for (int k0 = 0; k0 < D_MODEL; k0 += 32) {
        // distinct fragment variables -> staggered s_wait_loadcnt, no false reuse
        v16bf a0 = load_frag_a(xbf, D_MODEL, m0,      k0);
        v16bf a1 = load_frag_a(xbf, D_MODEL, m0 + 16, k0);
        v16bf b0 = load_frag_b(wbf, D_MODEL, n0,      k0);
        v16bf b1 = load_frag_b(wbf, D_MODEL, n0 + 16, k0);
        v16bf b2 = load_frag_b(wbf, D_MODEL, n0 + 32, k0);
        v16bf b3 = load_frag_b(wbf, D_MODEL, n0 + 48, k0);
        if (k0 + 32 < D_MODEL)
            __builtin_prefetch(xbf + (size_t)m0 * D_MODEL + k0 + 32, 0, 1);
        acc0[0] = WMMA_BF16(a0, b0, acc0[0]);
        acc1[0] = WMMA_BF16(a1, b0, acc1[0]);
        acc0[1] = WMMA_BF16(a0, b1, acc0[1]);
        acc1[1] = WMMA_BF16(a1, b1, acc1[1]);
        acc0[2] = WMMA_BF16(a0, b2, acc0[2]);
        acc1[2] = WMMA_BF16(a1, b2, acc1[2]);
        acc0[3] = WMMA_BF16(a0, b3, acc0[3]);
        acc1[3] = WMMA_BF16(a1, b3, acc1[3]);
    }

    int lane = threadIdx.x & 31;
    int c = lane & 15, hi = lane >> 4;
#pragma unroll
    for (int j = 0; j < 4; ++j) {
        int n   = n0 + j * 16 + c;
        float bv = bias[n];
        int w   = n / D_MODEL;          // 0=Q, 1=K, 2=V
        int rem = n - w * D_MODEL;
        int h   = rem >> 7;
        int hd  = rem & 127;
#pragma unroll
        for (int half = 0; half < 2; ++half) {
#pragma unroll
            for (int r = 0; r < 8; ++r) {
                int m = m0 + half * 16 + r + hi * 8;
                int b = m >> 11;            // / SEQ
                int s = m & (SEQ - 1);
                float av = half ? acc1[j][r] : acc0[j][r];
                uint16_t v = f32_to_bf16(av + bv);
                size_t bh = (size_t)b * NHEAD + h;
                if (w == 0)      Qbf[(bh * SEQ + s) * HDIM + hd]  = v;
                else if (w == 1) Kbf[(bh * SEQ + s) * HDIM + hd]  = v;
                else             Vtbf[(bh * HDIM + hd) * SEQ + s] = v;
            }
        }
    }
}

// ---------------- causal flash attention ----------------
// One wave owns 16 query rows of one (b,h); online softmax over 32-key blocks.
__global__ __launch_bounds__(256) void flash_attn_kernel(
    const uint16_t* __restrict__ Qbf, const uint16_t* __restrict__ Kbf,
    const uint16_t* __restrict__ Vtbf, uint16_t* __restrict__ Obf)
{
    __shared__ __align__(16) uint16_t lds_p[8][16][32];   // per-wave P tile (1 KB)
    int warp = threadIdx.x >> 5;
    int lane = threadIdx.x & 31;
    int c = lane & 15, hi = lane >> 4;

    int wave = blockIdx.x * 8 + warp;
    int bh = wave / (SEQ / 16);
    int qt = wave % (SEQ / 16);
    int q0 = qt * 16;

    const uint16_t* qbase = Qbf  + (size_t)bh * SEQ * HDIM;
    const uint16_t* kbase = Kbf  + (size_t)bh * SEQ * HDIM;
    const uint16_t* vbase = Vtbf + (size_t)bh * HDIM * SEQ;

    v16bf qf0 = load_frag_a(qbase, HDIM, q0, 0);
    v16bf qf1 = load_frag_a(qbase, HDIM, q0, 32);
    v16bf qf2 = load_frag_a(qbase, HDIM, q0, 64);
    v16bf qf3 = load_frag_a(qbase, HDIM, q0, 96);

    v8f o[8] = {};
    float mrow[8], lrow[8];
#pragma unroll
    for (int r = 0; r < 8; ++r) { mrow[r] = -3e38f; lrow[r] = 0.f; }

    const float scale = 0.08838834764831843f;   // 1/sqrt(128)
    int nkb = (q0 + 47) >> 5;                   // key blocks needed for causal range
    for (int kb = 0; kb < nkb; ++kb) {
        int k0 = kb * 32;

        // scores S = Q K^T for keys [k0, k0+32), as two 16x16 C tiles
        v8f sc[2];
#pragma unroll
        for (int half = 0; half < 2; ++half) {
            v8f s = {};
            s = WMMA_BF16(qf0, load_frag_b(kbase, HDIM, k0 + half * 16, 0),  s);
            s = WMMA_BF16(qf1, load_frag_b(kbase, HDIM, k0 + half * 16, 32), s);
            s = WMMA_BF16(qf2, load_frag_b(kbase, HDIM, k0 + half * 16, 64), s);
            s = WMMA_BF16(qf3, load_frag_b(kbase, HDIM, k0 + half * 16, 96), s);
            int key = k0 + half * 16 + c;
#pragma unroll
            for (int r = 0; r < 8; ++r) {
                float v  = s[r] * scale;
                int  row = q0 + r + hi * 8;
                s[r] = (key > row) ? -3e38f : v;    // causal mask
            }
            sc[half] = s;
        }

        // online softmax: per-row max/sum via 16-lane xor reductions
        float corr[8];
#pragma unroll
        for (int r = 0; r < 8; ++r) {
            float rm = fmaxf(sc[0][r], sc[1][r]);
            rm = fmaxf(rm, __shfl_xor(rm, 1, 32));
            rm = fmaxf(rm, __shfl_xor(rm, 2, 32));
            rm = fmaxf(rm, __shfl_xor(rm, 4, 32));
            rm = fmaxf(rm, __shfl_xor(rm, 8, 32));
            float mnew = fmaxf(mrow[r], rm);
            corr[r] = __expf(mrow[r] - mnew);
            float p0 = __expf(sc[0][r] - mnew);
            float p1 = __expf(sc[1][r] - mnew);
            sc[0][r] = p0; sc[1][r] = p1;
            float t = p0 + p1;
            t += __shfl_xor(t, 1, 32);
            t += __shfl_xor(t, 2, 32);
            t += __shfl_xor(t, 4, 32);
            t += __shfl_xor(t, 8, 32);
            lrow[r] = lrow[r] * corr[r] + t;
            mrow[r] = mnew;
        }
#pragma unroll
        for (int j = 0; j < 8; ++j)
#pragma unroll
            for (int r = 0; r < 8; ++r)
                o[j][r] *= corr[r];

        // repack P (C layout) -> A fragment layout via per-wave LDS tile
        uint16_t (*P)[32] = lds_p[warp];
#pragma unroll
        for (int half = 0; half < 2; ++half)
#pragma unroll
            for (int r = 0; r < 8; ++r)
                P[r + hi * 8][half * 16 + c] = f32_to_bf16(sc[half][r]);
        asm volatile("s_wait_dscnt 0" ::: "memory");   // LDS writes visible before reads
        const uint16_t* pp = &P[c][hi * 8];
        B32 pd;
        pd.lo = *reinterpret_cast<const uint4*>(pp);
        pd.hi = *reinterpret_cast<const uint4*>(pp + 16);
        v16bf pf = bitcast_frag(pd);

        // O += P @ V   (V^T layout makes B-fragments contiguous)
#pragma unroll
        for (int j = 0; j < 8; ++j)
            o[j] = WMMA_BF16(pf, load_frag_b(vbase, SEQ, j * 16, k0), o[j]);
    }

    // normalize and store attention output as (B,S,D) bf16
    float invl[8];
#pragma unroll
    for (int r = 0; r < 8; ++r) invl[r] = 1.0f / lrow[r];
    int b = bh >> 4, h = bh & 15;
#pragma unroll
    for (int j = 0; j < 8; ++j)
#pragma unroll
        for (int r = 0; r < 8; ++r) {
            int s = q0 + r + hi * 8;
            Obf[((size_t)(b * SEQ + s)) * D_MODEL + h * HDIM + j * 16 + c] =
                f32_to_bf16(o[j][r] * invl[r]);
        }
}

// ---------------- output projection: out = attn @ Wout^T + b (fp32 out) ----------------
// Wave tile: 32M x 64N.
__global__ __launch_bounds__(256) void out_gemm_kernel(
    const uint16_t* __restrict__ abf, const uint16_t* __restrict__ wbf,
    const float* __restrict__ bias, float* __restrict__ out)
{
    int wave = blockIdx.x * (blockDim.x >> 5) + (threadIdx.x >> 5);
    const int NT = D_MODEL / 64;
    int mt = wave / NT, nt = wave % NT;
    int m0 = mt * 32, n0 = nt * 64;

    v8f acc0[4] = {};
    v8f acc1[4] = {};
    for (int k0 = 0; k0 < D_MODEL; k0 += 32) {
        v16bf a0 = load_frag_a(abf, D_MODEL, m0,      k0);
        v16bf a1 = load_frag_a(abf, D_MODEL, m0 + 16, k0);
        v16bf b0 = load_frag_b(wbf, D_MODEL, n0,      k0);
        v16bf b1 = load_frag_b(wbf, D_MODEL, n0 + 16, k0);
        v16bf b2 = load_frag_b(wbf, D_MODEL, n0 + 32, k0);
        v16bf b3 = load_frag_b(wbf, D_MODEL, n0 + 48, k0);
        if (k0 + 32 < D_MODEL)
            __builtin_prefetch(abf + (size_t)m0 * D_MODEL + k0 + 32, 0, 1);
        acc0[0] = WMMA_BF16(a0, b0, acc0[0]);
        acc1[0] = WMMA_BF16(a1, b0, acc1[0]);
        acc0[1] = WMMA_BF16(a0, b1, acc0[1]);
        acc1[1] = WMMA_BF16(a1, b1, acc1[1]);
        acc0[2] = WMMA_BF16(a0, b2, acc0[2]);
        acc1[2] = WMMA_BF16(a1, b2, acc1[2]);
        acc0[3] = WMMA_BF16(a0, b3, acc0[3]);
        acc1[3] = WMMA_BF16(a1, b3, acc1[3]);
    }

    int lane = threadIdx.x & 31, c = lane & 15, hi = lane >> 4;
#pragma unroll
    for (int j = 0; j < 4; ++j) {
        int n = n0 + j * 16 + c;
        float bv = bias[n];
#pragma unroll
        for (int r = 0; r < 8; ++r) {
            int m = m0 + r + hi * 8;
            out[(size_t)m * D_MODEL + n] = acc0[j][r] + bv;
        }
#pragma unroll
        for (int r = 0; r < 8; ++r) {
            int m = m0 + 16 + r + hi * 8;
            out[(size_t)m * D_MODEL + n] = acc1[j][r] + bv;
        }
    }
}

// ---------------- host launch ----------------
extern "C" void kernel_launch(void* const* d_in, const int* in_sizes, int n_in,
                              void* d_out, int out_size, void* d_ws, size_t ws_size,
                              hipStream_t stream) {
    (void)in_sizes; (void)n_in; (void)out_size; (void)ws_size;
    const float* x     = (const float*)d_in[0];
    const float* w_qkv = (const float*)d_in[1];
    const float* b_qkv = (const float*)d_in[2];
    const float* w_out = (const float*)d_in[3];
    const float* b_out = (const float*)d_in[4];
    float* out = (float*)d_out;

    char* ws = (char*)d_ws;
    size_t off = 0;
    auto carve = [&](size_t elems) -> uint16_t* {
        uint16_t* p = (uint16_t*)(ws + off);
        off += (elems * sizeof(uint16_t) + 255) & ~(size_t)255;
        return p;
    };
    const size_t nx  = (size_t)BATCH * SEQ * D_MODEL;   // 8.4M elems
    const size_t nwq = (size_t)QKV_N * D_MODEL;         // 12.6M elems
    const size_t nwo = (size_t)D_MODEL * D_MODEL;       // 4.2M elems
    uint16_t* xbf  = carve(nx);
    uint16_t* wqbf = carve(nwq);
    uint16_t* wobf = carve(nwo);
    uint16_t* Qbf  = carve(nx);
    uint16_t* Kbf  = carve(nx);
    uint16_t* Vtbf = carve(nx);
    uint16_t* Obf  = carve(nx);

    auto cvt = [&](const float* s, uint16_t* d, size_t n) {
        int n4 = (int)(n / 4);
        cvt_f32_bf16_kernel<<<(n4 + 255) / 256, 256, 0, stream>>>(
            (const float4*)s, (ushort4*)d, n4);
    };
    cvt(x,     xbf,  nx);
    cvt(w_qkv, wqbf, nwq);
    cvt(w_out, wobf, nwo);

    int qkv_waves = (MROWS / 32) * (QKV_N / 64);        // 12288
    qkv_gemm_kernel<<<qkv_waves / 8, 256, 0, stream>>>(xbf, wqbf, b_qkv, Qbf, Kbf, Vtbf);

    int attn_waves = BATCH * NHEAD * (SEQ / 16);        // 4096
    flash_attn_kernel<<<attn_waves / 8, 256, 0, stream>>>(Qbf, Kbf, Vtbf, Obf);

    int out_waves = (MROWS / 32) * (D_MODEL / 64);      // 4096
    out_gemm_kernel<<<out_waves / 8, 256, 0, stream>>>(Obf, wobf, b_out, out);
}
